// GNN_86998857548354
// MI455X (gfx1250) — compile-verified, matrix operand order checked by reading, not run on hardware
//
#include <hip/hip_runtime.h>

// ---------------------------------------------------------------------------
// Model constants (from the reference)
// ---------------------------------------------------------------------------
#define BSZ   4096
#define NDN   20000
#define NPN   20000
#define EDN   640000
#define EPN   640000
#define DGO_D 2048
#define EPSV  1e-5f
#define SLOPE 0.01f

typedef __attribute__((ext_vector_type(16))) __bf16 v16bf;
typedef __attribute__((ext_vector_type(8)))  __bf16 v8bf;
typedef __attribute__((ext_vector_type(2)))  __bf16 v2bf;
typedef __attribute__((ext_vector_type(8)))  float  v8f;

union FragU { unsigned u[8]; v16bf v; };
union PairU { v2bf b; unsigned u; };

// ---------------------------------------------------------------------------
// Tiled bf16 WMMA GEMM: C[M,N] = A[M,K] @ B[K,N] (+bias), fp32 in/out.
// Block tile 128x128, K-step 32. 256 threads = 8 waves in a 2(M) x 4(N) grid;
// each wave owns a 64x32 output patch = 4x2 tiles of v_wmma_f32_16x16x32_bf16.
// A staged row-major bf16; B staged as packed (k,k+1) bf16 pairs, n-major,
// so LDS stores are b128-wide and fragment loads are b32 (ISA pair layout).
// ---------------------------------------------------------------------------
#define TM 128
#define TN 128
#define TK 32
#define AK (TK + 8)    // 40 bf16 = 80 B row stride (16B aligned)
#define BPN (TN + 4)   // 132 u32 = 528 B row stride (16B aligned)

__global__ __launch_bounds__(256) void gnn_wmma_gemm(
    const float* __restrict__ A, const float* __restrict__ B,
    const float* __restrict__ bias, float* __restrict__ C,
    int M, int N, int K)
{
    __shared__ __bf16   As[TM][AK];        // (m, k)
    __shared__ unsigned Bp[TK / 2][BPN];   // (kpair, n) packed bf16{k_even,k_odd}

    const int tid  = threadIdx.x;
    const int lane = tid & 31;
    const int wave = tid >> 5;          // 0..7
    const int wm   = wave >> 2;         // 0..1 -> +64*wm rows
    const int wn   = wave & 3;          // 0..3 -> +32*wn cols
    const int half = lane >> 4;         // hi/lo half of wave
    const int l16  = lane & 15;
    const int bM   = blockIdx.y * TM;
    const int bN   = blockIdx.x * TN;

    // loader assignments
    const int arow  = tid >> 1;         // 0..127
    const int acolb = (tid & 1) * 16;   // 0 | 16
    const int bkp   = tid >> 4;         // 0..15 (k-pair)
    const int bnb   = (tid & 15) * 8;   // 0..120

    const bool aInterior = (bM + TM <= M);
    const bool bInterior = (bN + TN <= N);

    v8f acc[4][2] = {};
    const int kb = half * 8;            // K-base for this wave half (ISA layout)

    for (int k0 = 0; k0 < K; k0 += TK) {
        const bool kfull = (k0 + TK <= K);

        // ---- stage A tile 128x32 ------------------------------------------
        if (aInterior && kfull) {
            const float4* ap =
                (const float4*)(A + (size_t)(bM + arow) * K + k0 + acolb);
            float4 x0 = ap[0], x1 = ap[1], x2 = ap[2], x3 = ap[3];
            v8bf lo, hi;
            lo[0] = (__bf16)x0.x; lo[1] = (__bf16)x0.y;
            lo[2] = (__bf16)x0.z; lo[3] = (__bf16)x0.w;
            lo[4] = (__bf16)x1.x; lo[5] = (__bf16)x1.y;
            lo[6] = (__bf16)x1.z; lo[7] = (__bf16)x1.w;
            hi[0] = (__bf16)x2.x; hi[1] = (__bf16)x2.y;
            hi[2] = (__bf16)x2.z; hi[3] = (__bf16)x2.w;
            hi[4] = (__bf16)x3.x; hi[5] = (__bf16)x3.y;
            hi[6] = (__bf16)x3.z; hi[7] = (__bf16)x3.w;
            *(v8bf*)&As[arow][acolb]     = lo;
            *(v8bf*)&As[arow][acolb + 8] = hi;
            if (k0 + TK < K)
                __builtin_prefetch(ap + 4 * (TK == 32 ? 2 : 2), 0, 3);
        } else {
            const int grow = bM + arow;
            #pragma unroll
            for (int j = 0; j < 16; ++j) {
                const int gk = k0 + acolb + j;
                float v = (grow < M && gk < K) ? A[(size_t)grow * K + gk] : 0.0f;
                As[arow][acolb + j] = (__bf16)v;
            }
        }

        // ---- stage B tile 32x128 as (k,k+1) pairs -------------------------
        if (bInterior && kfull) {
            const float* b0 = B + (size_t)(k0 + 2 * bkp) * N + bN + bnb;
            const float* b1 = b0 + N;
            float4 r0a = ((const float4*)b0)[0], r0b = ((const float4*)b0)[1];
            float4 r1a = ((const float4*)b1)[0], r1b = ((const float4*)b1)[1];
            const float ev[8] = {r0a.x, r0a.y, r0a.z, r0a.w,
                                 r0b.x, r0b.y, r0b.z, r0b.w};
            const float od[8] = {r1a.x, r1a.y, r1a.z, r1a.w,
                                 r1b.x, r1b.y, r1b.z, r1b.w};
            unsigned pk[8];
            #pragma unroll
            for (int j = 0; j < 8; ++j) {
                PairU t; t.b[0] = (__bf16)ev[j]; t.b[1] = (__bf16)od[j];
                pk[j] = t.u;
            }
            uint4 w0 = {pk[0], pk[1], pk[2], pk[3]};
            uint4 w1 = {pk[4], pk[5], pk[6], pk[7]};
            *(uint4*)&Bp[bkp][bnb]     = w0;
            *(uint4*)&Bp[bkp][bnb + 4] = w1;
            if (k0 + TK < K)
                __builtin_prefetch(b0 + (size_t)TK * N, 0, 3);
        } else {
            const int gk0v = k0 + 2 * bkp, gk1v = gk0v + 1;
            #pragma unroll
            for (int j = 0; j < 8; ++j) {
                const int gn = bN + bnb + j;
                float e = (gk0v < K && gn < N) ? B[(size_t)gk0v * N + gn] : 0.0f;
                float o = (gk1v < K && gn < N) ? B[(size_t)gk1v * N + gn] : 0.0f;
                PairU t; t.b[0] = (__bf16)e; t.b[1] = (__bf16)o;
                Bp[bkp][bnb + j] = t.u;
            }
        }
        __syncthreads();

        // ---- fragments (ISA 7.12.2 16-bit layouts) ------------------------
        FragU afr[4], bfr[2];
        #pragma unroll
        for (int ms = 0; ms < 4; ++ms) {
            const int m = wm * 64 + ms * 16 + l16;
            const unsigned* arp = (const unsigned*)&As[m][0];
            #pragma unroll
            for (int v = 0; v < 8; ++v) {
                const int k = (v < 4) ? (kb + 2 * v) : (kb + 16 + 2 * (v - 4));
                afr[ms].u[v] = arp[k >> 1];
            }
        }
        #pragma unroll
        for (int ns = 0; ns < 2; ++ns) {
            const int n = wn * 32 + ns * 16 + l16;
            const int kb2 = half * 4;
            #pragma unroll
            for (int v = 0; v < 8; ++v) {
                const int kp = (v < 4) ? (kb2 + v) : (kb2 + 8 + (v - 4));
                bfr[ns].u[v] = Bp[kp][n];
            }
        }

        #pragma unroll
        for (int ms = 0; ms < 4; ++ms)
            #pragma unroll
            for (int ns = 0; ns < 2; ++ns)
                acc[ms][ns] = __builtin_amdgcn_wmma_f32_16x16x32_bf16(
                    false, afr[ms].v, false, bfr[ns].v, (short)0, acc[ms][ns],
                    false, false);
        __syncthreads();
    }

    // ---- epilogue: C VGPR r -> M = r + half*8, N = l16 ----------------------
    #pragma unroll
    for (int ms = 0; ms < 4; ++ms) {
        #pragma unroll
        for (int ns = 0; ns < 2; ++ns) {
            const int col = bN + wn * 32 + ns * 16 + l16;
            #pragma unroll
            for (int r = 0; r < 8; ++r) {
                const int row = bM + wm * 64 + ms * 16 + half * 8 + r;
                if (row < M && col < N) {
                    float o = acc[ms][ns][r];
                    if (bias) o += bias[col];
                    C[(size_t)row * N + col] = o;
                }
            }
        }
    }
}

// ---------------------------------------------------------------------------
// BN stats (training mode): 64-way row-split partial sums + finalize.
// ---------------------------------------------------------------------------
__global__ void gnn_col_stats_part(const float* __restrict__ X, int R, int C,
                                   int rows_per_blk,
                                   float* __restrict__ sum, float* __restrict__ sum2)
{
    const int c = blockIdx.x * blockDim.x + threadIdx.x;
    if (c >= C) return;
    const int r0 = blockIdx.y * rows_per_blk;
    int r1 = r0 + rows_per_blk; if (r1 > R) r1 = R;
    float s = 0.f, s2 = 0.f;
    for (int r = r0; r < r1; ++r) {
        const float x = X[(size_t)r * C + c];
        s += x; s2 += x * x;
    }
    atomicAdd(&sum[c], s);
    atomicAdd(&sum2[c], s2);
}

__global__ void gnn_col_stats_fin(float* __restrict__ mean,
                                  float* __restrict__ var, int R, int C)
{
    const int c = blockIdx.x * blockDim.x + threadIdx.x;
    if (c >= C) return;
    const float m = mean[c] / (float)R;
    mean[c] = m;
    var[c]  = var[c] / (float)R - m * m;
}

// ---------------------------------------------------------------------------
// Fused BN-apply + activation + optional row gather + strided/offset store.
// act: 0 = none, 1 = relu, 2 = leaky-relu(0.01)
// ---------------------------------------------------------------------------
__global__ void gnn_bn_act(const float* __restrict__ X,
                           const float* __restrict__ mean,
                           const float* __restrict__ var,
                           const float* __restrict__ g,
                           const float* __restrict__ beta,
                           float* __restrict__ out,
                           int R, int C, int ostride, int ooff,
                           const int* __restrict__ idx, int act)
{
    const long long i = (long long)blockIdx.x * blockDim.x + threadIdx.x;
    if (i >= (long long)R * C) return;
    const int r = (int)(i / C);
    const int c = (int)(i % C);
    const int rs = idx ? idx[r] : r;
    float y = (X[(size_t)rs * C + c] - mean[c]) * rsqrtf(var[c] + EPSV) * g[c] + beta[c];
    if (act == 1)      y = fmaxf(y, 0.0f);
    else if (act == 2) y = (y > 0.0f) ? y : SLOPE * y;
    out[(size_t)r * ostride + ooff + c] = y;
}

__global__ void gnn_fill(float* __restrict__ p, float v, int n)
{
    const int i = blockIdx.x * blockDim.x + threadIdx.x;
    if (i < n) p[i] = v;
}

// deg[dst] += ew (ew==null -> 1.0); deg pre-filled with 1.0 (self loop).
__global__ void gnn_deg_scatter(const int* __restrict__ dst,
                                const float* __restrict__ ew,
                                float* __restrict__ deg, int E)
{
    const int e = blockIdx.x * blockDim.x + threadIdx.x;
    if (e < E) atomicAdd(&deg[dst[e]], ew ? ew[e] : 1.0f);
}

// agg = h/deg (self-loop term) + bias   (then edges accumulate on top)
__global__ void gnn_agg_init(const float* __restrict__ h,
                             const float* __restrict__ deg,
                             const float* __restrict__ bias,
                             float* __restrict__ agg, int Nn, int C)
{
    const long long i = (long long)blockIdx.x * blockDim.x + threadIdx.x;
    if (i >= (long long)Nn * C) return;
    const int r = (int)(i / C), c = (int)(i % C);
    agg[i] = h[i] / deg[r] + bias[c];
}

// One wave32 per edge: agg[dst] += h[src] * norm.  h & agg (82 MB each) both
// fit in the 192 MB L2, so the gather + atomic scatter stays on-die.
__global__ __launch_bounds__(256) void gnn_edge_agg(
    const int* __restrict__ src, const int* __restrict__ dst,
    const float* __restrict__ ew, const float* __restrict__ deg,
    const float* __restrict__ h, float* __restrict__ agg, int E, int C)
{
    const int e = blockIdx.x * 8 + (threadIdx.x >> 5);
    if (e >= E) return;
    const int lane = threadIdx.x & 31;
    const int s = src[e], d = dst[e];
    const float w = ew ? ew[e] : 1.0f;
    const float norm = rsqrtf(deg[s]) * w * rsqrtf(deg[d]);
    const float* hs = h + (size_t)s * C;
    float* ad = agg + (size_t)d * C;
    for (int c = lane; c < C; c += 32)
        atomicAdd(&ad[c], hs[c] * norm);
}

// y[r] = dot(H[r,:], W[:,0]) + b  (C = 256, block-per-row reduction)
__global__ void gnn_final_out(const float* __restrict__ H,
                              const float* __restrict__ W,
                              const float* __restrict__ b,
                              float* __restrict__ y, int C)
{
    __shared__ float red[256];
    const int r = blockIdx.x;
    float s = 0.f;
    for (int c = threadIdx.x; c < C; c += blockDim.x)
        s += H[(size_t)r * C + c] * W[c];
    red[threadIdx.x] = s;
    __syncthreads();
    for (int off = 128; off > 0; off >>= 1) {
        if ((int)threadIdx.x < off) red[threadIdx.x] += red[threadIdx.x + off];
        __syncthreads();
    }
    if (threadIdx.x == 0) y[r] = red[0] + b[0];
}

// ---------------------------------------------------------------------------
// Orchestration
// ---------------------------------------------------------------------------
extern "C" void kernel_launch(void* const* d_in, const int* in_sizes, int n_in,
                              void* d_out, int out_size, void* d_ws, size_t ws_size,
                              hipStream_t stream)
{
    (void)in_sizes; (void)n_in; (void)out_size; (void)ws_size;

    // ---- inputs (setup_inputs dict order, params flattened depth-first) ----
    const int*   d_index = (const int*)d_in[0];
    const int*   p_index = (const int*)d_in[1];
    const float* d_vecs  = (const float*)d_in[2];
    const float* p_embs  = (const float*)d_in[3];
    const float* d_ecfps = (const float*)d_in[4];
    const int*   d_ei    = (const int*)d_in[5];
    const float* d_ew    = (const float*)d_in[6];
    const float* p_gos   = (const float*)d_in[7];
    const int*   p_ei    = (const int*)d_in[8];
    const float* Wdv = (const float*)d_in[9],  *bdv = (const float*)d_in[10],
               *gdv = (const float*)d_in[11], *adv = (const float*)d_in[12];
    const float* Wpe = (const float*)d_in[13], *bpe = (const float*)d_in[14],
               *gpe = (const float*)d_in[15], *ape = (const float*)d_in[16];
    const float* Wec = (const float*)d_in[17], *bec = (const float*)d_in[18],
               *gec = (const float*)d_in[19], *aec = (const float*)d_in[20];
    const float* Wgo = (const float*)d_in[21], *bgo = (const float*)d_in[22],
               *ggo = (const float*)d_in[23], *ago = (const float*)d_in[24];
    const float* We1 = (const float*)d_in[25], *be1 = (const float*)d_in[26],
               *ge1 = (const float*)d_in[27], *ae1 = (const float*)d_in[28];
    const float* We2 = (const float*)d_in[29], *be2 = (const float*)d_in[30],
               *ge2 = (const float*)d_in[31], *ae2 = (const float*)d_in[32];
    const float* Wd1 = (const float*)d_in[33], *bd1 = (const float*)d_in[34],
               *gd1 = (const float*)d_in[35], *ad1 = (const float*)d_in[36];
    const float* Wd2 = (const float*)d_in[37], *bd2 = (const float*)d_in[38],
               *gd2 = (const float*)d_in[39], *ad2 = (const float*)d_in[40];
    const float* Wo1 = (const float*)d_in[41], *bo1 = (const float*)d_in[42],
               *go1 = (const float*)d_in[43], *ao1 = (const float*)d_in[44];
    const float* Wo2 = (const float*)d_in[45], *bo2 = (const float*)d_in[46];

    // ---- output layout: y[B] | decoded[B,3072] | feature[B,3072] -----------
    float* out_y    = (float*)d_out;
    float* decoded  = out_y + BSZ;
    float* feature  = decoded + (size_t)BSZ * 3072;

    // ---- workspace carve ----------------------------------------------------
    float* W = (float*)d_ws;
    size_t o = 0;
    float* hbuf   = W + o; o += (size_t)NDN * 1024;   // GCN h (reused)
    float* abuf   = W + o; o += (size_t)NDN * 1024;   // GCN agg (reused)
    float* fd     = W + o; o += (size_t)BSZ * 512;
    float* fp     = W + o; o += (size_t)BSZ * 512;
    float* enc1h  = W + o; o += (size_t)BSZ * 2048;
    float* enc2h  = W + o; o += (size_t)BSZ * 1024;   // "encoded"
    float* dec1h  = W + o; o += (size_t)BSZ * 2048;
    float* outh   = W + o; o += (size_t)BSZ * 256;
    float* deg    = W + o; o += NDN;
    float* mean   = W + o; o += 4096;
    float* var    = W + o; o += 4096;

    const dim3 blk(256);
    auto gemm = [&](const float* Am, const float* Bm, const float* bias,
                    float* Cm, int M, int N, int K) {
        dim3 grid((N + TN - 1) / TN, (M + TM - 1) / TM);
        gnn_wmma_gemm<<<grid, blk, 0, stream>>>(Am, Bm, bias, Cm, M, N, K);
    };
    auto stats = [&](const float* X, int R, int C) {
        const int SPLIT = 64;
        const int rpb = (R + SPLIT - 1) / SPLIT;
        gnn_fill<<<dim3((C + 255) / 256), blk, 0, stream>>>(mean, 0.0f, C);
        gnn_fill<<<dim3((C + 255) / 256), blk, 0, stream>>>(var,  0.0f, C);
        gnn_col_stats_part<<<dim3((C + 255) / 256, SPLIT), blk, 0, stream>>>(
            X, R, C, rpb, mean, var);
        gnn_col_stats_fin<<<dim3((C + 255) / 256), blk, 0, stream>>>(mean, var, R, C);
    };
    auto bnact = [&](const float* X, const float* g, const float* beta,
                     float* out, int R, int C, int ostride, int ooff,
                     const int* idx, int act) {
        long long tot = (long long)R * C;
        gnn_bn_act<<<dim3((unsigned)((tot + 255) / 256)), blk, 0, stream>>>(
            X, mean, var, g, beta, out, R, C, ostride, ooff, idx, act);
    };

    // ---- f_d = lrelu(BN(d_vecs @ Wdv + bdv)) -> feature[:, 0:512] ----------
    gemm(d_vecs, Wdv, bdv, fd, BSZ, 512, 300);
    stats(fd, BSZ, 512);
    bnact(fd, gdv, adv, feature, BSZ, 512, 3072, 0, nullptr, 2);

    // ---- f_p -> feature[:, 512:1024] ---------------------------------------
    gemm(p_embs, Wpe, bpe, fp, BSZ, 512, 1024);
    stats(fp, BSZ, 512);
    bnact(fp, gpe, ape, feature, BSZ, 512, 3072, 512, nullptr, 2);

    // ---- drug GCN -> feature[:, 1024:2048] (gathered by d_index) -----------
    gemm(d_ecfps, Wec, nullptr, hbuf, NDN, 1024, 1024);
    gnn_fill<<<dim3((NDN + 255) / 256), blk, 0, stream>>>(deg, 1.0f, NDN);
    gnn_deg_scatter<<<dim3((EDN + 255) / 256), blk, 0, stream>>>(d_ei + EDN, d_ew, deg, EDN);
    gnn_agg_init<<<dim3((unsigned)(((long long)NDN * 1024 + 255) / 256)), blk, 0, stream>>>(
        hbuf, deg, bec, abuf, NDN, 1024);
    gnn_edge_agg<<<dim3((EDN + 7) / 8), blk, 0, stream>>>(
        d_ei, d_ei + EDN, d_ew, deg, hbuf, abuf, EDN, 1024);
    stats(abuf, NDN, 1024);
    bnact(abuf, gec, aec, feature, BSZ, 1024, 3072, 1024, d_index, 2);

    // ---- protein GCN -> feature[:, 2048:3072] (gathered by p_index) --------
    gemm(p_gos, Wgo, nullptr, hbuf, NPN, 1024, DGO_D);
    gnn_fill<<<dim3((NPN + 255) / 256), blk, 0, stream>>>(deg, 1.0f, NPN);
    gnn_deg_scatter<<<dim3((EPN + 255) / 256), blk, 0, stream>>>(p_ei + EPN, nullptr, deg, EPN);
    gnn_agg_init<<<dim3((unsigned)(((long long)NPN * 1024 + 255) / 256)), blk, 0, stream>>>(
        hbuf, deg, bgo, abuf, NPN, 1024);
    gnn_edge_agg<<<dim3((EPN + 7) / 8), blk, 0, stream>>>(
        p_ei, p_ei + EPN, nullptr, deg, hbuf, abuf, EPN, 1024);
    stats(abuf, NPN, 1024);
    bnact(abuf, ggo, ago, feature, BSZ, 1024, 3072, 2048, p_index, 2);

    // ---- encoder ------------------------------------------------------------
    gemm(feature, We1, be1, enc1h, BSZ, 2048, 3072);
    stats(enc1h, BSZ, 2048);
    bnact(enc1h, ge1, ae1, enc1h, BSZ, 2048, 2048, 0, nullptr, 1);

    gemm(enc1h, We2, be2, enc2h, BSZ, 1024, 2048);
    stats(enc2h, BSZ, 1024);
    bnact(enc2h, ge2, ae2, enc2h, BSZ, 1024, 1024, 0, nullptr, 1);

    // ---- decoder -> decoded output -----------------------------------------
    gemm(enc2h, Wd1, bd1, dec1h, BSZ, 2048, 1024);
    stats(dec1h, BSZ, 2048);
    bnact(dec1h, gd1, ad1, dec1h, BSZ, 2048, 2048, 0, nullptr, 1);

    gemm(dec1h, Wd2, bd2, decoded, BSZ, 3072, 2048);
    stats(decoded, BSZ, 3072);
    bnact(decoded, gd2, ad2, decoded, BSZ, 3072, 3072, 0, nullptr, 1);

    // ---- head -> y ----------------------------------------------------------
    gemm(enc2h, Wo1, bo1, outh, BSZ, 256, 1024);
    stats(outh, BSZ, 256);
    bnact(outh, go1, ao1, outh, BSZ, 256, 256, 0, nullptr, 2);

    gnn_final_out<<<dim3(BSZ), blk, 0, stream>>>(outh, Wo2, bo2, out_y, 256);
}
